// Convolution_20968030339295
// MI455X (gfx1250) — matrix operand order repaired and checked by reading, not compiled, not run on hardware
//
#include <hip/hip_runtime.h>
#include <hip/hip_bf16.h>

typedef float  v2f   __attribute__((ext_vector_type(2)));
typedef float  v8f   __attribute__((ext_vector_type(8)));
typedef __bf16 v16bf __attribute__((ext_vector_type(16)));

#define WAVES_PER_BLOCK 4

__device__ __forceinline__ v8f wmma_f32_k4(v2f a, v2f b, v8f c) {
  return __builtin_amdgcn_wmma_f32_16x16x4_f32(false, a, false, b, (short)0, c, false, false);
}
__device__ __forceinline__ v8f wmma_bf16_k32(v16bf a, v16bf b, v8f c) {
  return __builtin_amdgcn_wmma_f32_16x16x32_bf16(false, a, false, b, (short)0, c, false, false);
}

// ---------------------------------------------------------------- zero ns/nv
__global__ void zero_kernel(float4* __restrict__ p, int n4) {
  int i = blockIdx.x * blockDim.x + threadIdx.x;
  int stride = gridDim.x * blockDim.x;
  float4 z = {0.f, 0.f, 0.f, 0.f};
  for (; i < n4; i += stride) p[i] = z;
}

// ---------------------------------------------------------------- node lin1
// s_lin = s @ (w1s/sqrt32); v_lin[:,u,i] channel-mix with w1v/sqrt32
__global__ void lin1_kernel(const float* __restrict__ node_input,
                            const float* __restrict__ w1s,
                            const float* __restrict__ w1v,
                            float* __restrict__ s_lin,
                            float* __restrict__ v_lin, int N) {
  __shared__ float Ws[32 * 32];
  __shared__ float Wv[32 * 32];
  const float sc = 0.17677669529663687f;  // 1/sqrt(32)
  for (int i = threadIdx.x; i < 1024; i += blockDim.x) {
    Ws[i] = w1s[i] * sc;
    Wv[i] = w1v[i] * sc;
  }
  __syncthreads();
  int lane = threadIdx.x & 31, wave = threadIdx.x >> 5;
  int n0 = (blockIdx.x * WAVES_PER_BLOCK + wave) * 16;
  if (n0 >= N) return;
  int row = lane & 15, half = lane >> 4;
  const float* __restrict__ Arow = node_input + (size_t)(n0 + row) * 128;

  // scalar channels (cols 0..31 of node_input)
  {
    v2f a[8];
#pragma unroll
    for (int kk = 0; kk < 8; ++kk) {
      int k = kk * 4 + half * 2;
      a[kk].x = Arow[k];
      a[kk].y = Arow[k + 1];
    }
#pragma unroll
    for (int nt = 0; nt < 2; ++nt) {
      int ncol = nt * 16 + row;
      v8f c = {};
#pragma unroll
      for (int kk = 0; kk < 8; ++kk) {
        int kb = kk * 4 + half * 2;
        v2f b;
        b.x = Ws[kb * 32 + ncol];
        b.y = Ws[(kb + 1) * 32 + ncol];
        c = wmma_f32_k4(a[kk], b, c);
      }
#pragma unroll
      for (int r = 0; r < 8; ++r)
        s_lin[(size_t)(n0 + r + half * 8) * 32 + ncol] = c[r];
    }
  }
  // vector channels: col = 32 + u*3 + comp
  for (int comp = 0; comp < 3; ++comp) {
    v2f a[8];
#pragma unroll
    for (int kk = 0; kk < 8; ++kk) {
      int k = kk * 4 + half * 2;
      a[kk].x = Arow[32 + k * 3 + comp];
      a[kk].y = Arow[32 + (k + 1) * 3 + comp];
    }
#pragma unroll
    for (int nt = 0; nt < 2; ++nt) {
      int ncol = nt * 16 + row;
      v8f c = {};
#pragma unroll
      for (int kk = 0; kk < 8; ++kk) {
        int kb = kk * 4 + half * 2;
        v2f b;
        b.x = Wv[kb * 32 + ncol];
        b.y = Wv[(kb + 1) * 32 + ncol];
        c = wmma_f32_k4(a[kk], b, c);
      }
#pragma unroll
      for (int r = 0; r < 8; ++r)
        v_lin[(size_t)(n0 + r + half * 8) * 96 + ncol * 3 + comp] = c[r];
    }
  }
}

// ---------------------------------------------------------------- edge kernel
// per wave: 16 edges. h (16x64) built per-lane in A-fragment layout (f32 VALU,
// K=8 dot + silu), then (16x64)@(64x128) via v_wmma_f32_16x16x32_bf16,
// consuming each 16-col C tile immediately into gather/scatter-atomic math.
__global__ void edge_kernel(const float* __restrict__ edge_scalars,
                            const int* __restrict__ edge_src,
                            const int* __restrict__ edge_dst,
                            const float* __restrict__ edge_attr,
                            const float* __restrict__ fc_w1,
                            const float* __restrict__ fc_w2,
                            const float* __restrict__ s_lin,
                            const float* __restrict__ v_lin,
                            float* __restrict__ ns,
                            float* __restrict__ nv, int E) {
  __shared__ float F1[8 * 64];
  __shared__ __align__(32) __bf16 F2[8][2][32][16];  // [ntile][kchunk][lane][elem]
  const float fc1s = 0.35355339059327373f;  // 1/sqrt(8)
  const float fc2s = 0.125f;                // 1/sqrt(64)
  for (int i = threadIdx.x; i < 512; i += blockDim.x) F1[i] = fc_w1[i] * fc1s;
  for (int i = threadIdx.x; i < 8192; i += blockDim.x) {
    int e = i & 15;
    int ln = (i >> 4) & 31;
    int c = (i >> 9) & 1;
    int t = i >> 10;
    // B-fragment layout (32x16 bf16): K = 32c + 16*(e/8) + 8*(lane>=16) + e%8, N = 16t + lane%16
    int K = c * 32 + (e >> 3) * 16 + (ln >> 4) * 8 + (e & 7);
    int Ncol = t * 16 + (ln & 15);
    F2[t][c][ln][e] = (__bf16)(fc_w2[K * 128 + Ncol] * fc2s);
  }
  __syncthreads();

  int lane = threadIdx.x & 31, wave = threadIdx.x >> 5;
  int e0 = (blockIdx.x * WAVES_PER_BLOCK + wave) * 16;
  if (e0 >= E) return;
  int row = lane & 15, half = lane >> 4;

  // edge meta held per lane (lanes m and m+16 both hold edge m)
  int em = e0 + row;
  int src_l = edge_src[em];
  int dst_l = edge_dst[em];
  const float* ea = edge_attr + (size_t)em * 4;
  float es_l = ea[0], evx_l = ea[1], evy_l = ea[2], evz_l = ea[3];

  // build A fragments: h[row][k] for exactly the k this lane's A slots need
  float x[8];
  const float* esc = edge_scalars + (size_t)em * 8;
#pragma unroll
  for (int j = 0; j < 8; ++j) x[j] = esc[j];
  v16bf a[2];
#pragma unroll
  for (int c = 0; c < 2; ++c) {
#pragma unroll
    for (int e = 0; e < 16; ++e) {
      // A-fragment layout (16x32 bf16): k = 32c + 16*(e/8) + 8*half + e%8
      int k = c * 32 + (e >> 3) * 16 + half * 8 + (e & 7);
      float acc = 0.f;
#pragma unroll
      for (int j = 0; j < 8; ++j) acc += x[j] * F1[j * 64 + k];
      float h = (acc / (1.f + __expf(-acc))) * 1.6791767923989418f;  // silu*norm
      a[c][e] = (__bf16)h;
    }
  }

  const float INV = 0.25f;                  // 1/sqrt(16)
  const float RS3 = 0.57735026918962576f;   // 1/sqrt(3)

  for (int t = 0; t < 8; ++t) {
    v8f acc = {};
#pragma unroll
    for (int c = 0; c < 2; ++c) {
      v16bf b = *(const v16bf*)&F2[t][c][lane][0];
      acc = wmma_bf16_k32(a[c], b, acc);
    }
    int g = t >> 1;               // 0:w0 1:w1 2:w2 3:w3
    int u = (t & 1) * 16 + row;   // channel 0..31
#pragma unroll
    for (int r = 0; r < 8; ++r) {
      int m = r + half * 8;       // edge row within tile
      float wv = acc[r];
      int srcm = __shfl(src_l, m, 32);
      int dstm = __shfl(dst_l, m, 32);
      float esm = __shfl(es_l, m, 32);
      float evx = __shfl(evx_l, m, 32);
      float evy = __shfl(evy_l, m, 32);
      float evz = __shfl(evz_l, m, 32);
      if (g == 0) {  // mid_s0 = w0*se*es
        float se = s_lin[(size_t)srcm * 32 + u];
        atomicAdd(&ns[(size_t)dstm * 64 + u], INV * wv * se * esm);
      } else if (g == 1) {  // mid_v0 = (w1*se) x ev
        float se = s_lin[(size_t)srcm * 32 + u];
        float f = INV * wv * se;
        float* p = &nv[(size_t)dstm * 192 + u * 3];
        atomicAdd(p + 0, f * evx);
        atomicAdd(p + 1, f * evy);
        atomicAdd(p + 2, f * evz);
      } else if (g == 2) {  // mid_v1 = (w2*es)*ve
        const float* vp = &v_lin[(size_t)srcm * 96 + u * 3];
        float f = INV * wv * esm;
        float* p = &nv[(size_t)dstm * 192 + (32 + u) * 3];
        atomicAdd(p + 0, f * vp[0]);
        atomicAdd(p + 1, f * vp[1]);
        atomicAdd(p + 2, f * vp[2]);
      } else {  // mid_s1 = w3 * (ve.ev)/sqrt3
        const float* vp = &v_lin[(size_t)srcm * 96 + u * 3];
        float dot = vp[0] * evx + vp[1] * evy + vp[2] * evz;
        atomicAdd(&ns[(size_t)dstm * 64 + 32 + u], INV * wv * dot * RS3);
      }
    }
  }
}

// ---------------------------------------------------------------- node lin2
__global__ void lin2_kernel(const float* __restrict__ ns,
                            const float* __restrict__ nv,
                            const float* __restrict__ w2s,
                            const float* __restrict__ w2v,
                            float* __restrict__ out, int N) {
  __shared__ float Ws[64 * 32];
  __shared__ float Wv[64 * 32];
  const float sc = 0.125f;  // 1/sqrt(64)
  for (int i = threadIdx.x; i < 2048; i += blockDim.x) {
    Ws[i] = w2s[i] * sc;
    Wv[i] = w2v[i] * sc;
  }
  __syncthreads();
  int lane = threadIdx.x & 31, wave = threadIdx.x >> 5;
  int n0 = (blockIdx.x * WAVES_PER_BLOCK + wave) * 16;
  if (n0 >= N) return;
  int row = lane & 15, half = lane >> 4;

  {  // out_s = ns @ w2s'
    const float* Arow = ns + (size_t)(n0 + row) * 64;
    v2f a[16];
#pragma unroll
    for (int kk = 0; kk < 16; ++kk) {
      int k = kk * 4 + half * 2;
      a[kk].x = Arow[k];
      a[kk].y = Arow[k + 1];
    }
#pragma unroll
    for (int nt = 0; nt < 2; ++nt) {
      int ncol = nt * 16 + row;
      v8f c = {};
#pragma unroll
      for (int kk = 0; kk < 16; ++kk) {
        int kb = kk * 4 + half * 2;
        v2f b;
        b.x = Ws[kb * 32 + ncol];
        b.y = Ws[(kb + 1) * 32 + ncol];
        c = wmma_f32_k4(a[kk], b, c);
      }
#pragma unroll
      for (int r = 0; r < 8; ++r)
        out[(size_t)(n0 + r + half * 8) * 128 + ncol] = c[r];
    }
  }
  for (int comp = 0; comp < 3; ++comp) {  // out_v channel-mix per component
    const float* Arow = nv + (size_t)(n0 + row) * 192;
    v2f a[16];
#pragma unroll
    for (int kk = 0; kk < 16; ++kk) {
      int k = kk * 4 + half * 2;
      a[kk].x = Arow[k * 3 + comp];
      a[kk].y = Arow[(k + 1) * 3 + comp];
    }
#pragma unroll
    for (int nt = 0; nt < 2; ++nt) {
      int ncol = nt * 16 + row;
      v8f c = {};
#pragma unroll
      for (int kk = 0; kk < 16; ++kk) {
        int kb = kk * 4 + half * 2;
        v2f b;
        b.x = Wv[kb * 32 + ncol];
        b.y = Wv[(kb + 1) * 32 + ncol];
        c = wmma_f32_k4(a[kk], b, c);
      }
#pragma unroll
      for (int r = 0; r < 8; ++r)
        out[(size_t)(n0 + r + half * 8) * 128 + 32 + ncol * 3 + comp] = c[r];
    }
  }
}

extern "C" void kernel_launch(void* const* d_in, const int* in_sizes, int n_in,
                              void* d_out, int out_size, void* d_ws, size_t ws_size,
                              hipStream_t stream) {
  const float* node_input   = (const float*)d_in[0];
  // d_in[1] node_attr: unused by the reference math
  const int*   edge_src     = (const int*)d_in[2];
  const int*   edge_dst     = (const int*)d_in[3];
  const float* edge_attr    = (const float*)d_in[4];
  const float* edge_scalars = (const float*)d_in[5];
  const float* w_lin1_s     = (const float*)d_in[6];
  const float* w_lin1_v     = (const float*)d_in[7];
  const float* fc_w1        = (const float*)d_in[8];
  const float* fc_w2        = (const float*)d_in[9];
  const float* w_lin2_s     = (const float*)d_in[10];
  const float* w_lin2_v     = (const float*)d_in[11];
  float* out = (float*)d_out;

  int N = in_sizes[0] / 128;
  int E = in_sizes[2];

  float* ws    = (float*)d_ws;
  float* s_lin = ws;                       // N*32
  float* v_lin = ws + (size_t)N * 32;      // N*96
  float* ns    = ws + (size_t)N * 128;     // N*64
  float* nv    = ws + (size_t)N * 192;     // N*192

  int n4 = (int)((size_t)N * 256 / 4);
  zero_kernel<<<2048, 256, 0, stream>>>((float4*)ns, n4);

  int nodeBlocks = ((N + 15) / 16 + WAVES_PER_BLOCK - 1) / WAVES_PER_BLOCK;
  lin1_kernel<<<nodeBlocks, 32 * WAVES_PER_BLOCK, 0, stream>>>(
      node_input, w_lin1_s, w_lin1_v, s_lin, v_lin, N);

  int edgeBlocks = ((E + 15) / 16 + WAVES_PER_BLOCK - 1) / WAVES_PER_BLOCK;
  edge_kernel<<<edgeBlocks, 32 * WAVES_PER_BLOCK, 0, stream>>>(
      edge_scalars, edge_src, edge_dst, edge_attr, fc_w1, fc_w2,
      s_lin, v_lin, ns, nv, E);

  lin2_kernel<<<nodeBlocks, 32 * WAVES_PER_BLOCK, 0, stream>>>(
      ns, nv, w_lin2_s, w_lin2_v, out, N);
}
